// VI_MMD_marg_45037027065945
// MI455X (gfx1250) — matrix operand order repaired and checked by reading, not compiled, not run on hardware
//
#include <hip/hip_runtime.h>
#include <hip/hip_bf16.h>
#include <math.h>

#define N_ROWS 8192
#define DDX    3072
#define DWZ    256
#define NTILE  512   // N_ROWS / 16

typedef __attribute__((ext_vector_type(16))) __bf16 v16bf;
typedef __attribute__((ext_vector_type(8)))  __bf16 v8bf;
typedef __attribute__((ext_vector_type(8)))  float  v8f;

__device__ __forceinline__ float nlp(float x, float mu, float lv) {
    const float LOG2PI = 1.8378770664093453f;
    float d = x - mu;
    return -0.5f * (lv + LOG2PI + d * d * expf(-lv));
}

__device__ __forceinline__ v16bf cat8(v8bf a, v8bf b) {
    return __builtin_shufflevector(a, b, 0,1,2,3,4,5,6,7,8,9,10,11,12,13,14,15);
}

// ---------------- init: zero the 7 accumulators ----------------
__global__ void init_kernel(float* __restrict__ acc) {
    if (threadIdx.x < 8) acc[threadIdx.x] = 0.0f;
}

// ---------------- per-row elementwise reductions + bf16 hi/lo staging ------
__global__ __launch_bounds__(256) void rowstats_kernel(
    const float* __restrict__ x, const int* __restrict__ y,
    const float* __restrict__ px_mu, const float* __restrict__ px_lv,
    const float* __restrict__ w, const float* __restrict__ z,
    const float* __restrict__ pw_mu, const float* __restrict__ pw_lv,
    const float* __restrict__ pz_mu, const float* __restrict__ pz_lv,
    const float* __restrict__ qw_mu, const float* __restrict__ qw_lv,
    const float* __restrict__ qz_mu, const float* __restrict__ qz_lv,
    const float* __restrict__ qy_logits,
    float* __restrict__ sq_w, float* __restrict__ sq_z, float* __restrict__ acc,
    __bf16* __restrict__ w_hi, __bf16* __restrict__ w_lo,
    __bf16* __restrict__ z_hi, __bf16* __restrict__ z_lo, int stage)
{
    const int row = blockIdx.x;
    const int t   = threadIdx.x;

    float lpx = 0.0f;
    {
        const float* xr = x     + (size_t)row * DDX;
        const float* mr = px_mu + (size_t)row * DDX;
        const float* lr = px_lv + (size_t)row * DDX;
        for (int k = t; k < DDX; k += 256) lpx += nlp(xr[k], mr[k], lr[k]);
    }
    const size_t o = (size_t)row * DWZ + t;   // t in [0,256) == DWZ
    const float wv = w[o], zv = z[o];
    const float klw = nlp(wv, qw_mu[o], qw_lv[o]) - nlp(wv, pw_mu[o], pw_lv[o]);
    const float klz = nlp(zv, qz_mu[o], qz_lv[o]) - nlp(zv, pz_mu[o], pz_lv[o]);
    float pm = -lpx + klw + klz;              // BX = BW = BZ = 1

    if (stage) {                              // one-time fp32 -> bf16 hi/lo split
        __bf16 h = (__bf16)wv;
        w_hi[o] = h; w_lo[o] = (__bf16)(wv - (float)h);
        h = (__bf16)zv;
        z_hi[o] = h; z_lo[o] = (__bf16)(zv - (float)h);
    }

    __shared__ float s0[256], s1[256], s2[256];
    s0[t] = pm; s1[t] = wv * wv; s2[t] = zv * zv;
    __syncthreads();
    for (int s = 128; s > 0; s >>= 1) {
        if (t < s) { s0[t] += s0[t + s]; s1[t] += s1[t + s]; s2[t] += s2[t + s]; }
        __syncthreads();
    }
    if (t == 0) {
        float lg = qy_logits[row];
        float ls = -log1pf(expf(-fabsf(lg)));
        float log_qy = ((y[row] != 0) ? fminf(lg, 0.0f) : fminf(-lg, 0.0f)) + ls;
        float m = s0[0] - log_qy;             // BY = 1
        atomicAdd(&acc[0], m * (1.0f / (float)N_ROWS));
        sq_w[row] = s1[0];
        sq_z[row] = s2[0];
    }
}

// epilogue helper: one 16x16 tile's contribution from one accumulator lane slot
__device__ __forceinline__ void tile_contrib(
    float dot, float sqi, float sqj, int mi, float mj0, float mj1,
    float inv, float& s00, float& s11, float& s01)
{
    float d2 = sqi + sqj - 2.0f * dot;
    d2 = fmaxf(d2, 0.0f);
    const float Kv = expf(-d2 * inv);
    const float mi0 = (mi == 0) ? 1.0f : 0.0f;
    const float mi1 = (mi == 1) ? 1.0f : 0.0f;
    s00 += mi0 * mj0 * Kv;
    s11 += mi1 * mj1 * Kv;
    s01 += mi0 * mj1 * Kv;
}

// ---------------- MMD v2: pre-split bf16 fragments + 2x2 register blocking
// block = 256 threads = 8 waves arranged 4(i) x 2(j); each wave owns a 32x32
// output patch (4 WMMA tiles). grid = (N/64, N/128).
__global__ __launch_bounds__(256) void mmd_bf16_kernel(
    const __bf16* __restrict__ hi, const __bf16* __restrict__ lo,
    const float* __restrict__ sq, const int* __restrict__ mask,
    float* __restrict__ acc3)
{
    const float inv = 1.0f / 65536.0f;        // 1 / (d*d*l*l), d=256, l=1
    const int t    = threadIdx.x;
    const int lane = t & 31;
    const int wave = t >> 5;
    const int wi   = wave & 3;
    const int wj   = wave >> 2;
    const int ibase = blockIdx.y * 128 + wi * 32;
    const int jbase = blockIdx.x * 64  + wj * 32;
    const int n    = lane & 15;
    const int half = lane >> 4;

    // A (16x32 bf16): lane = row M = lane&15; elem e -> k = half*8 + e (+8 if e>=8)
    // B (32x16 bf16): lane = col N = lane&15; elem e -> k = half*8 + e (+8 if e>=8)
    const __bf16* ahA = hi + (size_t)(ibase + n)      * DWZ;
    const __bf16* alA = lo + (size_t)(ibase + n)      * DWZ;
    const __bf16* ahB = hi + (size_t)(ibase + 16 + n) * DWZ;
    const __bf16* alB = lo + (size_t)(ibase + 16 + n) * DWZ;
    const __bf16* bhA = hi + (size_t)(jbase + n)      * DWZ;
    const __bf16* blA = lo + (size_t)(jbase + n)      * DWZ;
    const __bf16* bhB = hi + (size_t)(jbase + 16 + n) * DWZ;
    const __bf16* blB = lo + (size_t)(jbase + 16 + n) * DWZ;

    v8f acc00 = {}, acc01 = {}, acc10 = {}, acc11 = {};

    for (int kk = 0; kk < DWZ; kk += 32) {
        const int ka0 = kk + half * 8;
        const int ka1 = kk + 16 + half * 8;
        const int kb0 = kk + half * 16;

        v16bf ahiA = cat8(*(const v8bf*)(ahA + ka0), *(const v8bf*)(ahA + ka1));
        v16bf aloA = cat8(*(const v8bf*)(alA + ka0), *(const v8bf*)(alA + ka1));
        v16bf ahiB = cat8(*(const v8bf*)(ahB + ka0), *(const v8bf*)(ahB + ka1));
        v16bf aloB = cat8(*(const v8bf*)(alB + ka0), *(const v8bf*)(alB + ka1));
        v16bf bhiA = cat8(*(const v8bf*)(bhA + ka0), *(const v8bf*)(bhA + ka1));
        v16bf bloA = cat8(*(const v8bf*)(blA + ka0), *(const v8bf*)(blA + ka1));
        v16bf bhiB = cat8(*(const v8bf*)(bhB + ka0), *(const v8bf*)(bhB + ka1));
        v16bf bloB = cat8(*(const v8bf*)(blB + ka0), *(const v8bf*)(blB + ka1));

        // 3x bf16 split per tile: hi*hi + hi*lo + lo*hi (f32 accumulate)
        acc00 = __builtin_amdgcn_wmma_f32_16x16x32_bf16(false, ahiA, false, bhiA, (short)0, acc00, false, false);
        acc01 = __builtin_amdgcn_wmma_f32_16x16x32_bf16(false, ahiA, false, bhiB, (short)0, acc01, false, false);
        acc10 = __builtin_amdgcn_wmma_f32_16x16x32_bf16(false, ahiB, false, bhiA, (short)0, acc10, false, false);
        acc11 = __builtin_amdgcn_wmma_f32_16x16x32_bf16(false, ahiB, false, bhiB, (short)0, acc11, false, false);
        acc00 = __builtin_amdgcn_wmma_f32_16x16x32_bf16(false, ahiA, false, bloA, (short)0, acc00, false, false);
        acc01 = __builtin_amdgcn_wmma_f32_16x16x32_bf16(false, ahiA, false, bloB, (short)0, acc01, false, false);
        acc10 = __builtin_amdgcn_wmma_f32_16x16x32_bf16(false, ahiB, false, bloA, (short)0, acc10, false, false);
        acc11 = __builtin_amdgcn_wmma_f32_16x16x32_bf16(false, ahiB, false, bloB, (short)0, acc11, false, false);
        acc00 = __builtin_amdgcn_wmma_f32_16x16x32_bf16(false, aloA, false, bhiA, (short)0, acc00, false, false);
        acc01 = __builtin_amdgcn_wmma_f32_16x16x32_bf16(false, aloA, false, bhiB, (short)0, acc01, false, false);
        acc10 = __builtin_amdgcn_wmma_f32_16x16x32_bf16(false, aloB, false, bhiA, (short)0, acc10, false, false);
        acc11 = __builtin_amdgcn_wmma_f32_16x16x32_bf16(false, aloB, false, bhiB, (short)0, acc11, false, false);
    }

    // epilogue: dist2 -> exp -> masked sums over the 2x2 tile patch
    const int   mjAv = mask[jbase + n];
    const int   mjBv = mask[jbase + 16 + n];
    const float mjA0 = (mjAv == 0) ? 1.0f : 0.0f, mjA1 = (mjAv == 1) ? 1.0f : 0.0f;
    const float mjB0 = (mjBv == 0) ? 1.0f : 0.0f, mjB1 = (mjBv == 1) ? 1.0f : 0.0f;
    const float sqjA = sq[jbase + n];
    const float sqjB = sq[jbase + 16 + n];

    float s00 = 0.0f, s11 = 0.0f, s01 = 0.0f;
    #pragma unroll
    for (int r = 0; r < 8; ++r) {
        const int riA = ibase + r + 8 * half;   // C/D layout: M = r + 8*half
        const int riB = riA + 16;
        const float sqiA = sq[riA], sqiB = sq[riB];
        const int   miAv = mask[riA], miBv = mask[riB];
        tile_contrib(acc00[r], sqiA, sqjA, miAv, mjA0, mjA1, inv, s00, s11, s01);
        tile_contrib(acc01[r], sqiA, sqjB, miAv, mjB0, mjB1, inv, s00, s11, s01);
        tile_contrib(acc10[r], sqiB, sqjA, miBv, mjA0, mjA1, inv, s00, s11, s01);
        tile_contrib(acc11[r], sqiB, sqjB, miBv, mjB0, mjB1, inv, s00, s11, s01);
    }

    __shared__ float r0[256], r1[256], r2[256];
    r0[t] = s00; r1[t] = s11; r2[t] = s01;
    __syncthreads();
    for (int s = 128; s > 0; s >>= 1) {
        if (t < s) { r0[t] += r0[t + s]; r1[t] += r1[t + s]; r2[t] += r2[t + s]; }
        __syncthreads();
    }
    if (t == 0) {
        atomicAdd(&acc3[0], r0[0]);
        atomicAdd(&acc3[1], r1[0]);
        atomicAdd(&acc3[2], r2[0]);
    }
}

// ---------------- fallback MMD (on-the-fly conversion, small ws) -----------
__global__ __launch_bounds__(256) void mmd_fp32_kernel(
    const float* __restrict__ v, const float* __restrict__ sq,
    const int* __restrict__ mask, float* __restrict__ acc3)
{
    const float inv = 1.0f / 65536.0f;
    const int t    = threadIdx.x;
    const int lane = t & 31;
    const int wave = t >> 5;
    const int i0 = (blockIdx.y * 8 + wave) * 16;
    const int j0 = blockIdx.x * 16;
    const int n    = lane & 15;
    const int half = lane >> 4;

    const float* ar = v + (size_t)(i0 + n) * DWZ;
    const float* br = v + (size_t)(j0 + n) * DWZ;

    v8f accv = {};
    for (int kb = 0; kb < DWZ; kb += 32) {
        alignas(16) float af[16];
        alignas(16) float bf[16];
        *(float4*)(af + 0)  = *(const float4*)(ar + kb + half * 8);
        *(float4*)(af + 4)  = *(const float4*)(ar + kb + half * 8 + 4);
        *(float4*)(af + 8)  = *(const float4*)(ar + kb + 16 + half * 8);
        *(float4*)(af + 12) = *(const float4*)(ar + kb + 16 + half * 8 + 4);
        *(float4*)(bf + 0)  = *(const float4*)(br + kb + half * 8);
        *(float4*)(bf + 4)  = *(const float4*)(br + kb + half * 8 + 4);
        *(float4*)(bf + 8)  = *(const float4*)(br + kb + 16 + half * 8);
        *(float4*)(bf + 12) = *(const float4*)(br + kb + 16 + half * 8 + 4);

        v16bf ah, al, bh, bl;
        #pragma unroll
        for (int e = 0; e < 16; ++e) {
            __bf16 h = (__bf16)af[e];
            ah[e] = h; al[e] = (__bf16)(af[e] - (float)h);
            h = (__bf16)bf[e];
            bh[e] = h; bl[e] = (__bf16)(bf[e] - (float)h);
        }
        accv = __builtin_amdgcn_wmma_f32_16x16x32_bf16(false, ah, false, bh, (short)0, accv, false, false);
        accv = __builtin_amdgcn_wmma_f32_16x16x32_bf16(false, ah, false, bl, (short)0, accv, false, false);
        accv = __builtin_amdgcn_wmma_f32_16x16x32_bf16(false, al, false, bh, (short)0, accv, false, false);
    }

    const int   mj  = mask[j0 + n];
    const float mj0 = (mj == 0) ? 1.0f : 0.0f;
    const float mj1 = (mj == 1) ? 1.0f : 0.0f;
    const float sqj = sq[j0 + n];
    float s00 = 0.0f, s11 = 0.0f, s01 = 0.0f;
    #pragma unroll
    for (int r = 0; r < 8; ++r) {
        const int ri = i0 + r + 8 * half;
        tile_contrib(accv[r], sq[ri], sqj, mask[ri], mj0, mj1, inv, s00, s11, s01);
    }

    __shared__ float r0[256], r1[256], r2[256];
    r0[t] = s00; r1[t] = s11; r2[t] = s01;
    __syncthreads();
    for (int s = 128; s > 0; s >>= 1) {
        if (t < s) { r0[t] += r0[t + s]; r1[t] += r1[t + s]; r2[t] += r2[t + s]; }
        __syncthreads();
    }
    if (t == 0) {
        atomicAdd(&acc3[0], r0[0]);
        atomicAdd(&acc3[1], r1[0]);
        atomicAdd(&acc3[2], r2[0]);
    }
}

// ---------------- finalize: mask counts + combine ----------------
__global__ __launch_bounds__(256) void finalize_kernel(
    const int* __restrict__ c, const int* __restrict__ y,
    const float* __restrict__ acc, float* __restrict__ out)
{
    const int t = threadIdx.x;
    int c0 = 0, c1 = 0, yy0 = 0, yy1 = 0;
    for (int i = t; i < N_ROWS; i += 256) {
        c0  += (c[i] == 0); c1  += (c[i] == 1);
        yy0 += (y[i] == 0); yy1 += (y[i] == 1);
    }
    __shared__ int sc0[256], sc1[256], sy0[256], sy1[256];
    sc0[t] = c0; sc1[t] = c1; sy0[t] = yy0; sy1[t] = yy1;
    __syncthreads();
    for (int s = 128; s > 0; s >>= 1) {
        if (t < s) { sc0[t] += sc0[t+s]; sc1[t] += sc1[t+s]; sy0[t] += sy0[t+s]; sy1[t] += sy1[t+s]; }
        __syncthreads();
    }
    if (t == 0) {
        const float n0c = (float)sc0[0], n1c = (float)sc1[0];
        const float n0y = (float)sy0[0], n1y = (float)sy1[0];
        const float mmd_w = acc[1] / (n0c * n0c) + acc[2] / (n1c * n1c) - 2.0f * acc[3] / (n0c * n1c);
        const float mmd_z = acc[4] / (n0y * n0y) + acc[5] / (n1y * n1y) - 2.0f * acc[6] / (n0y * n1y);
        out[0] = acc[0] + 10.0f * mmd_w + 10.0f * mmd_z;  // BMZ=BMW=10
    }
}

extern "C" void kernel_launch(void* const* d_in, const int* in_sizes, int n_in,
                              void* d_out, int out_size, void* d_ws, size_t ws_size,
                              hipStream_t stream)
{
    const float* x     = (const float*)d_in[0];
    const int*   y     = (const int*)  d_in[1];
    const int*   c     = (const int*)  d_in[2];
    const float* px_mu = (const float*)d_in[3];
    const float* px_lv = (const float*)d_in[4];
    const float* w     = (const float*)d_in[5];
    const float* z     = (const float*)d_in[6];
    const float* pw_mu = (const float*)d_in[7];
    const float* pw_lv = (const float*)d_in[8];
    const float* pz_mu = (const float*)d_in[9];
    const float* pz_lv = (const float*)d_in[10];
    const float* qw_mu = (const float*)d_in[11];
    const float* qw_lv = (const float*)d_in[12];
    const float* qz_mu = (const float*)d_in[13];
    const float* qz_lv = (const float*)d_in[14];
    const float* qy    = (const float*)d_in[15];

    char*  wsb  = (char*)d_ws;
    float* sq_w = (float*)wsb;                     // N floats
    float* sq_z = (float*)(wsb + 32768);           // N floats
    float* acc  = (float*)(wsb + 65536);           // 8 floats
    __bf16* w_hi = (__bf16*)(wsb + 131072);        // 4 MB each below
    __bf16* w_lo = w_hi + (size_t)N_ROWS * DWZ;
    __bf16* z_hi = w_lo + (size_t)N_ROWS * DWZ;
    __bf16* z_lo = z_hi + (size_t)N_ROWS * DWZ;

    const size_t need = 131072 + 4ull * N_ROWS * DWZ * sizeof(__bf16);
    const int stage = (ws_size >= need) ? 1 : 0;

    init_kernel<<<1, 32, 0, stream>>>(acc);
    rowstats_kernel<<<N_ROWS, 256, 0, stream>>>(x, y, px_mu, px_lv, w, z,
                                                pw_mu, pw_lv, pz_mu, pz_lv,
                                                qw_mu, qw_lv, qz_mu, qz_lv,
                                                qy, sq_w, sq_z, acc,
                                                w_hi, w_lo, z_hi, z_lo, stage);
    if (stage) {
        dim3 g(N_ROWS / 64, N_ROWS / 128);
        mmd_bf16_kernel<<<g, 256, 0, stream>>>(w_hi, w_lo, sq_w, c, acc + 1);
        mmd_bf16_kernel<<<g, 256, 0, stream>>>(z_hi, z_lo, sq_z, y, acc + 4);
    } else {
        dim3 g(NTILE, NTILE / 8);
        mmd_fp32_kernel<<<g, 256, 0, stream>>>(w, sq_w, c, acc + 1);
        mmd_fp32_kernel<<<g, 256, 0, stream>>>(z, sq_z, y, acc + 4);
    }
    finalize_kernel<<<1, 256, 0, stream>>>(c, y, acc, (float*)d_out);
}
